// MambaWrapper_8658654069375
// MI455X (gfx1250) — matrix operand order repaired and checked by reading, not compiled, not run on hardware
//
#include <hip/hip_runtime.h>
#include <hip/hip_bf16.h>

// ---------------------------------------------------------------------------
// Mamba block forward for gfx1250 (MI455X).
//  - GEMMs: v_wmma_f32_16x16x32_bf16, bf16 operands / fp32 accumulate.
//  - Block-cooperative LDS staging through the CDNA5 async copy path
//    (global_load_async_to_lds_b128 + s_wait_asynccnt), double buffered.
//  - All fragment ds_loads issued before the 4 WMMAs so the XDL pipe runs
//    back-to-back after a single dscnt wait.
//  - Weights pre-transposed to bf16 [N x K]; activations produced in bf16.
//  - Selective scan: h[16] in registers per (batch, channel), fused gating.
// ---------------------------------------------------------------------------

typedef __attribute__((ext_vector_type(16))) __bf16 v16bf;
typedef __attribute__((ext_vector_type(8)))  __bf16 v8bf;
typedef __attribute__((ext_vector_type(8)))  float  v8f;

#define BATCH   2
#define SEQL    2048
#define DMODEL  1024
#define DSTATE  16
#define DCONV   4
#define DINNER  2048
#define DTRANK  64
#define NV2     32002
#define MTOT    (BATCH * SEQL)          // 4096 rows (B*L)
#define DBLW    (DTRANK + 2 * DSTATE)   // 96
#define NTILE   4                       // 16x16 tiles per wave along N
#define BM      64                      // block tile M
#define BN      128                     // block tile N
#define BK      32                      // K step

__device__ __forceinline__ float sigmoidf_(float x) { return 1.0f / (1.0f + __expf(-x)); }
__device__ __forceinline__ float siluf_(float x)    { return x * sigmoidf_(x); }
__device__ __forceinline__ float softplusf_(float x){ return (x > 20.0f) ? x : log1pf(__expf(x)); }

// 16B global -> LDS async copy (ASYNCcnt-tracked DMA, gfx1250)
__device__ __forceinline__ void async_cp16(const __bf16* g, __bf16* lds_generic) {
  uint32_t lds_off = (uint32_t)(uintptr_t)lds_generic;   // aperture low bits = LDS offset
  asm volatile("global_load_async_to_lds_b128 %0, %1, off"
               :: "v"(lds_off), "v"(g) : "memory");
}
__device__ __forceinline__ void wait_async3() {
  asm volatile("s_wait_asynccnt 0x3" ::: "memory");
}
__device__ __forceinline__ void wait_async0() {
  asm volatile("s_wait_asynccnt 0x0" ::: "memory");
}

// ---- Weight transpose + bf16 convert: W[K x N] -> Wt[N x K] ----------------
__global__ __launch_bounds__(256) void k_transpose_bf16(const float* __restrict__ W,
                                                        __bf16* __restrict__ Wt,
                                                        int K, int N) {
  __shared__ __bf16 tile[32][33];
  const int n0 = blockIdx.x * 32, k0 = blockIdx.y * 32;
  const int tx = threadIdx.x & 31, ty = threadIdx.x >> 5;   // 8 rows per pass
#pragma unroll
  for (int r = ty; r < 32; r += 8) {
    int k = k0 + r, n = n0 + tx;
    tile[r][tx] = (k < K && n < N) ? (__bf16)W[(size_t)k * N + n] : (__bf16)0.0f;
  }
  __syncthreads();
#pragma unroll
  for (int r = ty; r < 32; r += 8) {
    int n = n0 + r, k = k0 + tx;
    if (n < N && k < K) Wt[(size_t)n * K + k] = tile[tx][r];
  }
}

// ---- LDS-staged bf16 WMMA GEMM --------------------------------------------
// 256 threads = 8 waves as 4(M) x 2(N); each wave: 16(M) x 64(N) = 4 wmma
// accumulators sharing one A fragment.  Block tile 64 x 128, K-step 32.
// Tiles staged in LDS via async DMA (double buffered).
//   A  : bf16 [M x lda]      Bt : bf16 [N x K]
// mode 0: C0 fp32 [M x ldc]          mode 1: C1 bf16 [M x ldc]
// mode 2: both                       mode 3: transposed+bias fp32 out[b,n,l]
__global__ __launch_bounds__(256) void k_gemm(const __bf16* __restrict__ A,
                                              const __bf16* __restrict__ Bt,
                                              float* __restrict__ C0,
                                              __bf16* __restrict__ C1,
                                              const float* __restrict__ bias,
                                              int M, int N, int K,
                                              int lda, int ldc,
                                              int mode, int batchL) {
  __shared__ alignas(128) __bf16 sA[2][BM][BK];   // 2 x 4 KB
  __shared__ alignas(128) __bf16 sB[2][BN][BK];   // 2 x 8 KB
  __shared__ float smem[8][16][17];               // per-wave transpose (mode 3)

  const int tid  = threadIdx.x;
  const int wid  = tid >> 5;
  const int lane = tid & 31;
  const int half = lane >> 4;
  const int ln   = lane & 15;
  const int wm   = wid >> 1;
  const int wn   = wid & 1;

  const int m0b = blockIdx.y * BM;
  const int n0b = blockIdx.x * BN;
  const int m0  = m0b + wm * 16;
  const int nw  = n0b + wn * 64;                  // wave's N base (4 tiles)

  // ---- cooperative async-copy assignments (per thread: 1xA + 2xB 16B) ----
  const int r_a = tid >> 2;                       // A row 0..63
  const int c_a = tid & 3;                        // 16B chunk within 64B row
  const int r_b = tid >> 1;                       // B row 0..127
  const int c_b = tid & 1;                        // 32B chunk within 64B row
  int browg = n0b + r_b; if (browg >= N) browg = N - 1;      // clamp (stores guarded)
  const __bf16* gA = A  + (size_t)(m0b + r_a) * lda + c_a * 8;
  const __bf16* gB = Bt + (size_t)browg * K + c_b * 16;

  auto issue = [&](int buf, int kk) {
    async_cp16(gA + kk,     &sA[buf][r_a][c_a * 8]);
    async_cp16(gB + kk,     &sB[buf][r_b][c_b * 16]);
    async_cp16(gB + kk + 8, &sB[buf][r_b][c_b * 16 + 8]);
  };

  v8f zero = {};
  v8f acc[NTILE];
#pragma unroll
  for (int t = 0; t < NTILE; ++t) acc[t] = zero;

  const int arow_l = wm * 16 + ln;                // A fragment row in sA

  auto compute = [&](int cur) {
    // Stage ALL fragment reads first, then run the 4 WMMAs back-to-back.
    // A fragment: elements 0..15 = K 16*half..+15 (contiguous)
    v8bf a0 = *(const v8bf*)&sA[cur][arow_l][16 * half];
    v8bf a1 = *(const v8bf*)&sA[cur][arow_l][16 * half + 8];
    v16bf bf[NTILE];
#pragma unroll
    for (int t = 0; t < NTILE; ++t)   // B fragment: 16 contiguous bf16
      bf[t] = *(const v16bf*)&sB[cur][wn * 64 + 16 * t + ln][16 * half];
    v16bf a;
#pragma unroll
    for (int e = 0; e < 8; ++e) { a[e] = a0[e]; a[8 + e] = a1[e]; }
#pragma unroll
    for (int t = 0; t < NTILE; ++t)
      acc[t] = __builtin_amdgcn_wmma_f32_16x16x32_bf16(false, a, false, bf[t],
                                                       (short)0, acc[t], false, false);
  };

  // ---- async double-buffered main loop (last iteration peeled) ----
  const int nIt = K >> 5;                         // K is a multiple of 32
  issue(0, 0);
  int it = 0;
  for (; it < nIt - 1; ++it) {
    const int cur = it & 1;
    issue(cur ^ 1, (it + 1) * BK);
    wait_async3();                                // cur buffer's 3 copies done
    __syncthreads();
    compute(cur);
    __syncthreads();                              // all waves done reading cur
  }
  wait_async0();
  __syncthreads();
  compute(it & 1);

  // ---- epilogues ----  C layout: VGPR j -> M = m0 + 8*half + j, N = col
#pragma unroll
  for (int t = 0; t < NTILE; ++t) {
    const int  col   = nw + 16 * t + ln;
    const bool colok = (col < N);
    if (mode == 3) {
      float bv = colok ? bias[col] : 0.0f;
#pragma unroll
      for (int j = 0; j < 8; ++j)
        smem[wid][half * 8 + j][ln] = acc[t][j] + bv;   // smem[local_l][local_v]
      const int bb = m0 / batchL;
      const int l0 = m0 % batchL;                       // tile never straddles batch
#pragma unroll
      for (int it2 = 0; it2 < 8; ++it2) {
        int vi = half + 2 * it2;                        // local v (0..15)
        int v  = nw + 16 * t + vi;
        if (v < N)
          C0[((size_t)bb * N + v) * batchL + (l0 + ln)] = smem[wid][ln][vi];
      }
    } else if (colok) {
#pragma unroll
      for (int j = 0; j < 8; ++j) {
        int   r = m0 + half * 8 + j;
        float v = acc[t][j];
        if (mode == 0 || mode == 2) C0[(size_t)r * ldc + col] = v;
        if (mode == 1 || mode == 2) C1[(size_t)r * ldc + col] = (__bf16)v;
      }
    }
  }
}

// ---- Embedding gather -> bf16 ---------------------------------------------
__global__ __launch_bounds__(256) void k_embed(const int* __restrict__ x,
                                               const float* __restrict__ emb,
                                               __bf16* __restrict__ tokb) {
  const int m = blockIdx.x;
  const float* src = emb  + (size_t)x[m] * DMODEL;
  __bf16*      dst = tokb + (size_t)m * DMODEL;
  for (int d = threadIdx.x; d < DMODEL; d += 256) dst[d] = (__bf16)src[d];
}

// ---- Causal depthwise conv1d + SiLU (writes f32 for scan + bf16 for GEMM) --
__global__ __launch_bounds__(256) void k_conv_silu(const float* __restrict__ xz,
                                                   const float* __restrict__ conv_w,
                                                   const float* __restrict__ conv_b,
                                                   float* __restrict__ xc,
                                                   __bf16* __restrict__ xcb) {
  size_t idx = (size_t)blockIdx.x * 256 + threadIdx.x;
  if (idx >= (size_t)MTOT * DINNER) return;
  const int d = (int)(idx % DINNER);
  const int m = (int)(idx / DINNER);
  const int l = m % SEQL;
  float acc = conv_b[d];
#pragma unroll
  for (int j = 0; j < DCONV; ++j) {
    int t = l - (DCONV - 1) + j;
    if (t >= 0)
      acc += xz[(size_t)(m - (DCONV - 1) + j) * (2 * DINNER) + d] * conv_w[d * DCONV + j];
  }
  float v = siluf_(acc);
  xc[idx]  = v;
  xcb[idx] = (__bf16)v;
}

// ---- dt = softplus(dt_lin + b_dt) (in place, f32) --------------------------
__global__ __launch_bounds__(256) void k_softplus_bias(float* __restrict__ dtb,
                                                       const float* __restrict__ b_dt) {
  size_t idx = (size_t)blockIdx.x * 256 + threadIdx.x;
  if (idx >= (size_t)MTOT * DINNER) return;
  const int d = (int)(idx % DINNER);
  dtb[idx] = softplusf_(dtb[idx] + b_dt[d]);
}

// ---- Selective scan + gating (bf16 output feeds W_out GEMM) ----------------
__global__ __launch_bounds__(256) void k_scan(const float* __restrict__ dt,
                                              const float* __restrict__ dbl,
                                              const float* __restrict__ xc,
                                              const float* __restrict__ xz,
                                              const float* __restrict__ A_log,
                                              const float* __restrict__ D_skip,
                                              __bf16* __restrict__ yb) {
  const int gid = blockIdx.x * 256 + threadIdx.x;   // 0 .. BATCH*DINNER-1
  const int b = gid / DINNER;
  const int d = gid % DINNER;

  float Ad[DSTATE], h[DSTATE];
#pragma unroll
  for (int n = 0; n < DSTATE; ++n) {
    Ad[n] = -__expf(A_log[d * DSTATE + n]);
    h[n]  = 0.0f;
  }
  const float Dk = D_skip[d];

  for (int t = 0; t < SEQL; ++t) {
    const size_t m  = (size_t)b * SEQL + t;
    const float dtv = dt[m * DINNER + d];
    const float xv  = xc[m * DINNER + d];
    const float dx  = dtv * xv;
    const float* bc = dbl + m * DBLW;
    float acc = 0.0f;
#pragma unroll
    for (int n = 0; n < DSTATE; ++n) {
      float dA = __expf(dtv * Ad[n]);
      h[n] = dA * h[n] + dx * bc[DTRANK + n];
      acc += h[n] * bc[DTRANK + DSTATE + n];
    }
    const float zv = xz[m * (2 * DINNER) + DINNER + d];
    yb[m * DINNER + d] = (__bf16)((acc + xv * Dk) * siluf_(zv));
  }
}

// ---------------------------------------------------------------------------
static inline char* carve(char*& p, size_t bytes) {
  char* r = p;
  p += (bytes + 255) & ~(size_t)255;
  return r;
}

extern "C" void kernel_launch(void* const* d_in, const int* in_sizes, int n_in,
                              void* d_out, int out_size, void* d_ws, size_t ws_size,
                              hipStream_t stream) {
  const int*   x      = (const int*)  d_in[0];
  const float* emb    = (const float*)d_in[1];
  const float* W_in   = (const float*)d_in[2];
  const float* conv_w = (const float*)d_in[3];
  const float* conv_b = (const float*)d_in[4];
  const float* W_x    = (const float*)d_in[5];
  const float* W_dt   = (const float*)d_in[6];
  const float* b_dt   = (const float*)d_in[7];
  const float* A_log  = (const float*)d_in[8];
  const float* D_skip = (const float*)d_in[9];
  const float* W_out  = (const float*)d_in[10];
  const float* W_head = (const float*)d_in[11];
  const float* b_head = (const float*)d_in[12];
  float* out = (float*)d_out;

  // ---- workspace carve-up ----
  char* p = (char*)d_ws;
  __bf16* tokb  = (__bf16*)carve(p, (size_t)MTOT * DMODEL * 2);
  float*  xz    = (float*) carve(p, (size_t)MTOT * 2 * DINNER * 4);
  float*  xc    = (float*) carve(p, (size_t)MTOT * DINNER * 4);
  __bf16* xcb   = (__bf16*)carve(p, (size_t)MTOT * DINNER * 2);
  float*  dbl   = (float*) carve(p, (size_t)MTOT * DBLW * 4);
  __bf16* dblb  = (__bf16*)carve(p, (size_t)MTOT * DBLW * 2);
  float*  dtb   = (float*) carve(p, (size_t)MTOT * DINNER * 4);
  __bf16* ybb   = (__bf16*)carve(p, (size_t)MTOT * DINNER * 2);
  __bf16* moutb = (__bf16*)carve(p, (size_t)MTOT * DMODEL * 2);
  __bf16* WinT  = (__bf16*)carve(p, (size_t)(2 * DINNER) * DMODEL * 2); // [4096 x 1024]
  __bf16* WxT   = (__bf16*)carve(p, (size_t)DBLW * DINNER * 2);         // [96 x 2048]
  __bf16* WdtT  = (__bf16*)carve(p, (size_t)DINNER * DTRANK * 2);       // [2048 x 64]
  __bf16* WoutT = (__bf16*)carve(p, (size_t)DMODEL * DINNER * 2);       // [1024 x 2048]
  __bf16* WhT   = (__bf16*)carve(p, (size_t)NV2 * DMODEL * 2);          // [32002 x 1024]

  const dim3 blk(256);
  const size_t nMD = (size_t)MTOT * DINNER;
  auto tgrid = [](int K, int N) { return dim3((N + 31) / 32, (K + 31) / 32); };
  auto ggrid = [](int M, int N) { return dim3((N + BN - 1) / BN, M / BM); };

  // ---- weight transposes (f32 [KxN] -> bf16 [NxK]) ----
  k_transpose_bf16<<<tgrid(DMODEL, 2 * DINNER), blk, 0, stream>>>(W_in,  WinT,  DMODEL, 2 * DINNER);
  k_transpose_bf16<<<tgrid(DINNER, DBLW),       blk, 0, stream>>>(W_x,   WxT,   DINNER, DBLW);
  k_transpose_bf16<<<tgrid(DTRANK, DINNER),     blk, 0, stream>>>(W_dt,  WdtT,  DTRANK, DINNER);
  k_transpose_bf16<<<tgrid(DINNER, DMODEL),     blk, 0, stream>>>(W_out, WoutT, DINNER, DMODEL);
  k_transpose_bf16<<<tgrid(DMODEL, NV2),        blk, 0, stream>>>(W_head, WhT,  DMODEL, NV2);

  // 1) token embedding gather -> bf16
  k_embed<<<MTOT, blk, 0, stream>>>(x, emb, tokb);

  // 2) xz = tok @ W_in   [4096 x 4096], K=1024  (f32 out)
  k_gemm<<<ggrid(MTOT, 2 * DINNER), blk, 0, stream>>>(
      tokb, WinT, xz, nullptr, nullptr,
      MTOT, 2 * DINNER, DMODEL, DMODEL, 2 * DINNER, 0, 0);

  // 3) causal depthwise conv + silu -> xc (f32 + bf16)
  k_conv_silu<<<(unsigned)((nMD + 255) / 256), blk, 0, stream>>>(xz, conv_w, conv_b, xc, xcb);

  // 4) dbl = xc @ W_x    [4096 x 96], K=2048   (f32 + bf16 out)
  k_gemm<<<ggrid(MTOT, DBLW), blk, 0, stream>>>(
      xcb, WxT, dbl, dblb, nullptr,
      MTOT, DBLW, DINNER, DINNER, DBLW, 2, 0);

  // 5) dt_lin = dt_r @ W_dt  [4096 x 2048], K=64 (A = dblb[:, :64], lda=96)
  k_gemm<<<ggrid(MTOT, DINNER), blk, 0, stream>>>(
      dblb, WdtT, dtb, nullptr, nullptr,
      MTOT, DINNER, DTRANK, DBLW, DINNER, 0, 0);

  // 6) dt = softplus(dt_lin + b_dt)
  k_softplus_bias<<<(unsigned)((nMD + 255) / 256), blk, 0, stream>>>(dtb, b_dt);

  // 7) selective scan + D-skip + silu(z) gating -> bf16
  k_scan<<<(BATCH * DINNER) / 256, blk, 0, stream>>>(dtb, dbl, xc, xz, A_log, D_skip, ybb);

  // 8) m_out = y @ W_out  [4096 x 1024], K=2048  (bf16 out, feeds head GEMM)
  k_gemm<<<ggrid(MTOT, DMODEL), blk, 0, stream>>>(
      ybb, WoutT, nullptr, moutb, nullptr,
      MTOT, DMODEL, DINNER, DINNER, DMODEL, 1, 0);

  // 9) logits = m_out @ W_head + b_head -> transposed store out[B, V2, L]
  k_gemm<<<ggrid(MTOT, NV2), blk, 0, stream>>>(
      moutb, WhT, out, nullptr, b_head,
      MTOT, NV2, DMODEL, DMODEL, NV2, 3, SEQL);
}
